// MultiHeadAttention_20899310862617
// MI455X (gfx1250) — matrix operand order repaired and checked by reading, not compiled, not run on hardware
//
#include <hip/hip_runtime.h>
#include <hip/hip_bf16.h>

// ---------------------------------------------------------------------------
// Masked multi-head attention for MI455X (gfx1250), f16 WMMA with f32 accum.
//   H=8 heads, d_in=512, dk=dv=64, nq=g=4096, d_embed=512
// Workspace layout (f16, ~18 MB): WqT WkT WvT WoT | Qh Kh Vt Hh
// Mask tiles are streamed to LDS with the Tensor Data Mover (TDM),
// double-buffered on TENSORcnt so the DMA overlaps WMMA + softmax.
// ---------------------------------------------------------------------------

#define H    8
#define DIN  512
#define DK   64
#define DV   64
#define DE   512
#define NQ   4096
#define G    4096
#define NORMF 0.125f   // 1/sqrt(64)

typedef __attribute__((ext_vector_type(16))) _Float16 v16h;
typedef __attribute__((ext_vector_type(8)))  _Float16 v8h;
typedef __attribute__((ext_vector_type(8)))  float    v8f;
typedef __attribute__((ext_vector_type(4)))  unsigned int u32x4;
typedef __attribute__((ext_vector_type(8)))  unsigned int u32x8;

#define WMMA_F16(a, b, c) \
  __builtin_amdgcn_wmma_f32_16x16x32_f16(false, (a), false, (b), (short)0, (c), false, false)

static __device__ __forceinline__ v8f zero8() {
  v8f z = {0.f, 0.f, 0.f, 0.f, 0.f, 0.f, 0.f, 0.f};
  return z;
}

static __device__ __forceinline__ v16h pack16(v8h lo, v8h hi) {
  v16h r;
#pragma unroll
  for (int i = 0; i < 8; ++i) { r[i] = lo[i]; r[i + 8] = hi[i]; }
  return r;
}

// A-fragment (16x32 f16). lane&15 = row M, lane>>4 selects K-half.
// elements 0..15 hold K = c0..c0+15,
// where c0 = kk + (lane>>4)*16.
static __device__ __forceinline__ v16h load_a_f16(const _Float16* rowp, int c0) {
  v8h lo = *(const v8h*)(rowp + c0);
  v8h hi = *(const v8h*)(rowp + c0 + 8);
  return pack16(lo, hi);
}

// B-fragment (32x16 f16). lane&15 = col N, lane>>4 = K-half; 16 contiguous K.
static __device__ __forceinline__ v16h load_b_f16(const _Float16* p) {
  v8h lo = *(const v8h*)(p);
  v8h hi = *(const v8h*)(p + 8);
  return pack16(lo, hi);
}

// A-fragment straight from f32 memory (converted to f16).
static __device__ __forceinline__ v16h load_a_f32(const float* rowp, int c0) {
  float4 f0 = *(const float4*)(rowp + c0);
  float4 f1 = *(const float4*)(rowp + c0 + 4);
  float4 f2 = *(const float4*)(rowp + c0 + 8);
  float4 f3 = *(const float4*)(rowp + c0 + 12);
  v16h r;
  r[0]  = (_Float16)f0.x; r[1]  = (_Float16)f0.y; r[2]  = (_Float16)f0.z; r[3]  = (_Float16)f0.w;
  r[4]  = (_Float16)f1.x; r[5]  = (_Float16)f1.y; r[6]  = (_Float16)f1.z; r[7]  = (_Float16)f1.w;
  r[8]  = (_Float16)f2.x; r[9]  = (_Float16)f2.y; r[10] = (_Float16)f2.z; r[11] = (_Float16)f2.w;
  r[12] = (_Float16)f3.x; r[13] = (_Float16)f3.y; r[14] = (_Float16)f3.z; r[15] = (_Float16)f3.w;
  return r;
}

// ---------------------------------------------------------------------------
// TDM: load a 16(row) x 32(col) int32 tile of the 4096x4096 mask into LDS.
// D# per CDNA5 ISA:
//   group0: count=1 | lds_addr | global_addr[56:0] | type=2
//   group1: data_size=4B, tensor_dim0=tensor_dim1=4096, tile_dim0=32,
//           tile_dim1=16, tensor_dim0_stride=4096
//   groups 2/3: zero (2D tensor)
// Tile lands row-major contiguous in LDS: lds[y*128 + x*4].
// ---------------------------------------------------------------------------
static __device__ __forceinline__ void tdm_load_mask_tile(const int* gaddr,
                                                          unsigned lds_off) {
  unsigned long long ga = (unsigned long long)(size_t)gaddr;
  u32x4 g0;
  g0[0] = 1u;                                  // count=1 (valid), user mode
  g0[1] = lds_off;                             // lds_addr (bytes)
  g0[2] = (unsigned)(ga & 0xffffffffu);        // global_addr[31:0]
  g0[3] = (unsigned)((ga >> 32) & 0x1ffffffu)  // global_addr[56:32]
          | (2u << 30);                        // type=2 ("image")
  u32x8 g1;
  g1[0] = (2u << 16);        // workgroup_mask=0, data_size=2 (4 bytes)
  g1[1] = (4096u << 16);     // tensor_dim0[15:0] in [31:16]
  g1[2] = (4096u << 16);     // tensor_dim0[31:16]=0 | tensor_dim1[15:0] << 16
  g1[3] = (32u << 16);       // tensor_dim1[31:16]=0 | tile_dim0=32
  g1[4] = 16u;               // tile_dim1=16, tile_dim2=0
  g1[5] = 4096u;             // tensor_dim0_stride[31:0]
  g1[6] = 0u;                // stride[47:32]=0 | tensor_dim1_stride[15:0]=0
  g1[7] = 0u;
  u32x4 gz = {0u, 0u, 0u, 0u};
  asm volatile("tensor_load_to_lds %0, %1, %2, %3"
               :: "s"(g0), "s"(g1), "s"(gz), "s"(gz)
               : "memory");
}

// ---------------------------------------------------------------------------
// Kernel 0: convert & transpose weights to f16 so WMMA B-fragments are
// contiguous along the reduction dim.
// ---------------------------------------------------------------------------
__global__ void mha_wconvert(const float* __restrict__ Wq, const float* __restrict__ Wk,
                             const float* __restrict__ Wv, const float* __restrict__ Wo,
                             _Float16* __restrict__ WqT, _Float16* __restrict__ WkT,
                             _Float16* __restrict__ WvT, _Float16* __restrict__ WoT) {
  int i = blockIdx.x * blockDim.x + threadIdx.x;  // 0 .. 262143
  int which = blockIdx.y;
  if (which < 3) {
    const float* src = (which == 0) ? Wq : (which == 1 ? Wk : Wv);
    _Float16* dst = (which == 0) ? WqT : (which == 1 ? WkT : WvT);
    int h = i / (DIN * DK);
    int rem = i - h * DIN * DK;
    int k = rem / DK;
    int n = rem - k * DK;
    dst[((size_t)h * DK + n) * DIN + k] = (_Float16)src[i];
  } else {
    int h = i / (DK * DE);
    int rem = i - h * DK * DE;
    int k = rem / DE;
    int e = rem - k * DE;
    WoT[((size_t)h * DE + e) * DK + k] = (_Float16)Wo[i];
  }
}

// ---------------------------------------------------------------------------
// Kernel 1: QKV projection. One wave per 16(M) x 64(N) output tile.
//   grid = (NQ/16, H, 3)   z: 0=Q(from q), 1=K(from h), 2=V(from h)
// Q,K stored row-major [head][row][d] f16; V stored transposed Vt[head][v][g].
// ---------------------------------------------------------------------------
__global__ void __launch_bounds__(32) mha_qkv(const float* __restrict__ qin,
                                              const float* __restrict__ hin,
                                              const _Float16* __restrict__ WqT,
                                              const _Float16* __restrict__ WkT,
                                              const _Float16* __restrict__ WvT,
                                              _Float16* __restrict__ Qh,
                                              _Float16* __restrict__ Kh,
                                              _Float16* __restrict__ Vt) {
  const int lane  = threadIdx.x & 31;
  const int cn    = lane & 15;
  const int chalf = lane >> 4;
  const int tile  = blockIdx.x;
  const int head  = blockIdx.y;
  const int which = blockIdx.z;

  const float*    src = (which == 0) ? qin : hin;
  const _Float16* W   = (which == 0) ? WqT : (which == 1 ? WkT : WvT);
  const int row0 = tile * 16;
  const float* arow = src + (size_t)(row0 + cn) * DIN;

  v8f acc[4];
#pragma unroll
  for (int t = 0; t < 4; ++t) acc[t] = zero8();

  for (int kk = 0; kk < DIN; kk += 32) {
    v16h a = load_a_f32(arow, kk + chalf * 16);
#pragma unroll
    for (int t = 0; t < 4; ++t) {
      const _Float16* bp =
          W + ((size_t)head * DK + t * 16 + cn) * DIN + kk + chalf * 16;
      v16h b = load_b_f16(bp);
      acc[t] = WMMA_F16(a, b, acc[t]);
    }
  }

  if (which == 2) {
    // Vt[head][v][g]: for a lane the 8 rows (m) are contiguous in g -> v8h store
#pragma unroll
    for (int t = 0; t < 4; ++t) {
      v8h o;
#pragma unroll
      for (int r = 0; r < 8; ++r) o[r] = (_Float16)acc[t][r];
      *(v8h*)(Vt + ((size_t)head * DV + t * 16 + cn) * G + row0 + chalf * 8) = o;
    }
  } else {
    _Float16* dst = (which == 0) ? Qh : Kh;
#pragma unroll
    for (int t = 0; t < 4; ++t)
#pragma unroll
      for (int r = 0; r < 8; ++r) {
        int m = r + 8 * chalf;
        dst[((size_t)head * NQ + row0 + m) * DK + t * 16 + cn] = (_Float16)acc[t][r];
      }
  }
}

// ---------------------------------------------------------------------------
// Kernel 2: flash attention. Block = 256 threads = 8 waves; wave w = head w.
// All heads share one 16-query tile so the 67 MB mask is read from HBM once.
// Wave 0 streams mask tiles via TDM into a double-buffered LDS tile; the DMA
// for tile i+1 overlaps WMMA/softmax on tile i (one barrier per step).
//   grid = NQ/16
// ---------------------------------------------------------------------------
__global__ void __launch_bounds__(256, 1)
mha_attention(const _Float16* __restrict__ Qh, const _Float16* __restrict__ Kh,
              const _Float16* __restrict__ Vt, const int* __restrict__ mask,
              _Float16* __restrict__ Hh) {
  __shared__ __align__(16) int      mt[2][16][32];    // [buf][row][key_local]
  __shared__ __align__(16) _Float16 plds[H][16][32];  // per-head P tile

  const int lane  = threadIdx.x & 31;
  const int w     = threadIdx.x >> 5;  // head
  const int cn    = lane & 15;
  const int chalf = lane >> 4;
  const int q0    = blockIdx.x * 16;

  const _Float16* Qbase = Qh + ((size_t)w * NQ + q0) * DK;
  // Q fragments for head-dim chunks 0..31 / 32..63, loaded once.
  v16h aq0 = load_a_f16(Qbase + (size_t)cn * DK, chalf * 16);
  v16h aq1 = load_a_f16(Qbase + (size_t)cn * DK, 32 + chalf * 16);

  v8f O[4];
#pragma unroll
  for (int t = 0; t < 4; ++t) O[t] = zero8();
  float mrun[8], lrun[8];
#pragma unroll
  for (int r = 0; r < 8; ++r) { mrun[r] = -__builtin_inff(); lrun[r] = 0.f; }

  const int* mbase = mask + (size_t)q0 * G;

  // prime the pipeline: TDM fetch of tile 0 into buffer 0
  if (w == 0) tdm_load_mask_tile(mbase, (unsigned)(size_t)&mt[0][0][0]);

  for (int g0 = 0, it = 0; g0 < G; g0 += 32, ++it) {
    const int cur = it & 1;
    if (w == 0) __builtin_amdgcn_s_wait_tensorcnt(0);  // tile `it` has landed
    __syncthreads();  // publish tile `it`; prior consumers of buf[1-cur] done
    if (w == 0 && g0 + 32 < G)
      tdm_load_mask_tile(mbase + g0 + 32, (unsigned)(size_t)&mt[1 - cur][0][0]);

    const _Float16* Kbase = Kh + ((size_t)w * NQ + g0) * DK;
    if (g0 + 32 < G) __builtin_prefetch(Kbase + (size_t)32 * DK, 0, 1);

    // scores: two 16x16 key sub-tiles, head-dim reduced in two K=32 chunks
    v8f s0 = zero8(), s1 = zero8();
    {
      v16h b0 = load_b_f16(Kbase + (size_t)cn * DK + chalf * 16);
      v16h b1 = load_b_f16(Kbase + (size_t)cn * DK + 32 + chalf * 16);
      s0 = WMMA_F16(aq0, b0, s0);
      s0 = WMMA_F16(aq1, b1, s0);
      v16h b2 = load_b_f16(Kbase + (size_t)(cn + 16) * DK + chalf * 16);
      v16h b3 = load_b_f16(Kbase + (size_t)(cn + 16) * DK + 32 + chalf * 16);
      s1 = WMMA_F16(aq0, b2, s1);
      s1 = WMMA_F16(aq1, b3, s1);
    }

    // mask bits for this lane's column (key) across its 8 rows (TDM tile is
    // row-major in LDS)
    int mlo[8], mhi[8];
#pragma unroll
    for (int r = 0; r < 8; ++r) {
      mlo[r] = mt[cur][chalf * 8 + r][cn];
      mhi[r] = mt[cur][chalf * 8 + r][16 + cn];
    }

#pragma unroll
    for (int r = 0; r < 8; ++r) {
      float x0 = s0[r] * NORMF;
      float x1 = s1[r] * NORMF;
      // faithful to reference: masked OR exactly-zero compat -> -inf
      x0 = (mlo[r] != 0 && x0 != 0.f) ? x0 : -__builtin_inff();
      x1 = (mhi[r] != 0 && x1 != 0.f) ? x1 : -__builtin_inff();

      float cm = fmaxf(x0, x1);
#pragma unroll
      for (int off = 8; off >= 1; off >>= 1)
        cm = fmaxf(cm, __shfl_xor(cm, off, 32));  // row spans 16 lanes of a half

      float mnew = fmaxf(mrun[r], cm);
      float p0, p1, scale;
      if (mnew == -__builtin_inff()) {
        p0 = 0.f; p1 = 0.f; scale = 0.f;
      } else {
        p0 = __expf(x0 - mnew);
        p1 = __expf(x1 - mnew);
        scale = (mrun[r] == -__builtin_inff()) ? 0.f : __expf(mrun[r] - mnew);
      }
      float rs = p0 + p1;
#pragma unroll
      for (int off = 8; off >= 1; off >>= 1) rs += __shfl_xor(rs, off, 32);

      lrun[r] = lrun[r] * scale + rs;
      mrun[r] = mnew;
#pragma unroll
      for (int t = 0; t < 4; ++t) O[t][r] *= scale;

      int m = r + 8 * chalf;
      plds[w][m][cn]      = (_Float16)p0;   // C layout -> row-major P in LDS
      plds[w][m][16 + cn] = (_Float16)p1;
    }

    // wave is lockstep & LDS in-order per wave; fence against reordering
    asm volatile("s_wait_dscnt 0x0" ::: "memory");

    // P as A-fragment (16 rows x 32 keys), V as B-fragments (keys contiguous)
    v16h ap = load_a_f16(&plds[w][cn][0], chalf * 16);
    const _Float16* Vbase = Vt + (size_t)w * DV * G + g0;
#pragma unroll
    for (int t = 0; t < 4; ++t) {
      v16h bv = load_b_f16(Vbase + (size_t)(t * 16 + cn) * G + chalf * 16);
      O[t] = WMMA_F16(ap, bv, O[t]);
    }
  }

  // normalize and store heads (post-softmax mask multiply is a no-op: p==0 there)
  float inv[8];
#pragma unroll
  for (int r = 0; r < 8; ++r) inv[r] = (lrun[r] > 0.f) ? 1.f / lrun[r] : 0.f;
  _Float16* Hbase = Hh + ((size_t)w * NQ + q0) * DV;
#pragma unroll
  for (int t = 0; t < 4; ++t)
#pragma unroll
    for (int r = 0; r < 8; ++r) {
      int m = r + 8 * chalf;
      Hbase[(size_t)m * DV + t * 16 + cn] = (_Float16)(O[t][r] * inv[r]);
    }
}

// ---------------------------------------------------------------------------
// Kernel 3: output projection  out[q][e] = sum_h heads[h][q][:] @ Wo[h][:][e]
// One wave per 16(q) x 64(e) tile; reduction over 8 heads x 64 head-dim.
//   grid = (NQ/16, DE/64)
// ---------------------------------------------------------------------------
__global__ void __launch_bounds__(32) mha_outproj(const _Float16* __restrict__ Hh,
                                                  const _Float16* __restrict__ WoT,
                                                  float* __restrict__ out) {
  const int lane  = threadIdx.x & 31;
  const int cn    = lane & 15;
  const int chalf = lane >> 4;
  const int q0 = blockIdx.x * 16;
  const int e0 = blockIdx.y * 64;

  v8f acc[4];
#pragma unroll
  for (int t = 0; t < 4; ++t) acc[t] = zero8();

  for (int h = 0; h < H; ++h) {
    const _Float16* Abase = Hh + ((size_t)h * NQ + q0) * DV + (size_t)cn * DV;
    const _Float16* Bbase = WoT + (size_t)h * DE * DK;
#pragma unroll
    for (int kk = 0; kk < DK; kk += 32) {
      v16h a = load_a_f16(Abase, kk + chalf * 16);
#pragma unroll
      for (int t = 0; t < 4; ++t) {
        v16h b = load_b_f16(Bbase + (size_t)(e0 + t * 16 + cn) * DK + kk + chalf * 16);
        acc[t] = WMMA_F16(a, b, acc[t]);
      }
    }
  }

#pragma unroll
  for (int t = 0; t < 4; ++t)
#pragma unroll
    for (int r = 0; r < 8; ++r) {
      int m = r + 8 * chalf;
      out[(size_t)(q0 + m) * DE + e0 + t * 16 + cn] = acc[t][r];
    }
}

// ---------------------------------------------------------------------------
extern "C" void kernel_launch(void* const* d_in, const int* in_sizes, int n_in,
                              void* d_out, int out_size, void* d_ws, size_t ws_size,
                              hipStream_t stream) {
  const float* q    = (const float*)d_in[0];
  const float* h    = (const float*)d_in[1];
  const int*   mask = (const int*)d_in[2];
  const float* Wq   = (const float*)d_in[3];
  const float* Wk   = (const float*)d_in[4];
  const float* Wv   = (const float*)d_in[5];
  const float* Wo   = (const float*)d_in[6];

  // workspace carve-up (f16 elements); total ~18.9 MB
  _Float16* WqT = (_Float16*)d_ws;
  _Float16* WkT = WqT + (size_t)H * DK * DIN;
  _Float16* WvT = WkT + (size_t)H * DK * DIN;
  _Float16* WoT = WvT + (size_t)H * DK * DIN;
  _Float16* Qh  = WoT + (size_t)H * DE * DK;
  _Float16* Kh  = Qh + (size_t)H * NQ * DK;
  _Float16* Vt  = Kh + (size_t)H * G * DK;
  _Float16* Hh  = Vt + (size_t)H * DV * G;

  mha_wconvert<<<dim3((H * DIN * DK) / 256, 4), dim3(256), 0, stream>>>(
      Wq, Wk, Wv, Wo, WqT, WkT, WvT, WoT);

  mha_qkv<<<dim3(NQ / 16, H, 3), dim3(32), 0, stream>>>(
      q, h, WqT, WkT, WvT, Qh, Kh, Vt);

  mha_attention<<<dim3(NQ / 16), dim3(256), 0, stream>>>(Qh, Kh, Vt, mask, Hh);

  mha_outproj<<<dim3(NQ / 16, DE / 64), dim3(32), 0, stream>>>(
      Hh, WoT, (float*)d_out);
}
